// SumThenReg3RpeSelfAttention_34050500723068
// MI455X (gfx1250) — matrix operand order repaired and checked by reading, not compiled, not run on hardware
//
#include <hip/hip_runtime.h>
#include <hip/hip_bf16.h>

// ---------------------------------------------------------------------------
// CDNA5 (gfx1250) implementation: all contractions via v_wmma_f32_16x16x32_bf16
// (bf16 operands, fp32 accumulation). wave32 everywhere.
// ---------------------------------------------------------------------------

typedef __attribute__((ext_vector_type(16))) __bf16 v16bf;
typedef __attribute__((ext_vector_type(8)))  float  v8f;

constexpr int B_ = 2, H_ = 8, D_ = 64, E_ = 512, R_ = 2048, S_ = 128;
constexpr int N_ = R_ * B_;              // 4096 token rows
constexpr float SCALE_ = 0.125f;         // D^-0.5 = 64^-0.5

// ---- bf16 helpers (bit-level, round-to-nearest-even) ----------------------
__device__ __forceinline__ unsigned short f2bf(float f) {
  unsigned int u = __float_as_uint(f);
  u += 0x7FFFu + ((u >> 16) & 1u);
  return (unsigned short)(u >> 16);
}

union Frag16 { v16bf bf; unsigned int u[8]; };

// ---- WMMA fragment loaders, contiguous k = half*16 + s for A and B (wave32) -----------
// A matrix 16x32 bf16: lane(0-15)->M=lane, VGPR j holds K = half*16+2j,(+1);
// VGPR 4+j holds K = half*16+8+2j. half = lane>>4.
__device__ __forceinline__ void load_a_frag(const unsigned short* __restrict__ src,
                                            int ld, int lane, unsigned int* out) {
  int half = lane >> 4, m = lane & 15;
  const unsigned short* row = src + (size_t)m * ld;
#pragma unroll
  for (int j = 0; j < 4; ++j) {
    int k0 = half * 16 + 2 * j;
    out[j]     = (unsigned)row[k0] | ((unsigned)row[k0 + 1] << 16);
    int k1 = half * 16 + 8 + 2 * j;
    out[4 + j] = (unsigned)row[k1] | ((unsigned)row[k1 + 1] << 16);
  }
}

// B matrix 32x16 bf16 where B[kk][n] = src[n*ld + kk]  (i.e. B = M^T for a
// row-major M of shape [16 x >=32]); lanes 0-15 hold kk=0..15, lanes 16-31
// hold kk=16..31; VGPR j holds kk = half*16 + 2j,(+1); column n = lane&15.
__device__ __forceinline__ void load_bT_frag(const unsigned short* __restrict__ src,
                                             int ld, int lane, unsigned int* out) {
  int half = lane >> 4, n = lane & 15;
  const unsigned short* row = src + (size_t)n * ld;
#pragma unroll
  for (int j = 0; j < 8; ++j) {
    int kk = half * 16 + 2 * j;
    out[j] = (unsigned)row[kk] | ((unsigned)row[kk + 1] << 16);
  }
}

// B matrix 32x16 bf16 where B[kk][n] = src[kk*ld + n]  (natural orientation).
__device__ __forceinline__ void load_b_frag(const unsigned short* __restrict__ src,
                                            int ld, int lane, unsigned int* out) {
  int half = lane >> 4, n = lane & 15;
#pragma unroll
  for (int j = 0; j < 8; ++j) {
    int kk = half * 16 + 2 * j;
    out[j] = (unsigned)src[(size_t)kk * ld + n] |
             ((unsigned)src[(size_t)(kk + 1) * ld + n] << 16);
  }
}

// ---------------------------------------------------------------------------
// Elementwise fp32 -> bf16 conversion
// ---------------------------------------------------------------------------
__global__ void __launch_bounds__(256)
f32_to_bf16_kernel(const float* __restrict__ src, unsigned short* __restrict__ dst, int n) {
  int i = blockIdx.x * 256 + threadIdx.x;
  if (i < n) dst[i] = f2bf(src[i]);
}

// ---------------------------------------------------------------------------
// Summary-query gather: sum_q[b,h,s,d] = embed[ids[b,s], h*64+d] * SCALE (bf16)
// ---------------------------------------------------------------------------
__global__ void __launch_bounds__(256)
gather_sumq_kernel(const float* __restrict__ emb, const int* __restrict__ ids,
                   unsigned short* __restrict__ outQ) {
  int idx = blockIdx.x * 256 + threadIdx.x;
  if (idx >= B_ * H_ * S_ * D_) return;
  int d = idx & 63;
  int t = idx >> 6;
  int s = t % S_; t /= S_;
  int h = t % H_;
  int b = t / H_;
  int tok = ids[b * S_ + s];
  float v = emb[(size_t)tok * E_ + (h << 6) + d] * SCALE_;
  outQ[idx] = f2bf(v);
}

// ---------------------------------------------------------------------------
// Generic WMMA GEMM:  C[N,O] = X[N,I] (bf16) @ W[O,I]^T (bf16) + bias
// One 16(M) x 64(O) tile per wave: the A fragment is loaded once per k-step
// and amortized over 4 WMMAs (4 independent accumulation chains also hide the
// WMMA->WMMA RAW hazard). 8 waves (256 threads) per block.
//   outF  : optional fp32 [N,O]
//   outB  : optional bf16; headMode=0 -> [N,O]; headMode=1 -> [B,H,N/B,64]
//           with row n = r*B+b, col o = h*64+d; bf16 copy scaled by bscale.
// ---------------------------------------------------------------------------
__global__ void __launch_bounds__(256)
gemm_xwT_kernel(const unsigned short* __restrict__ X,
                const unsigned short* __restrict__ W,
                const float* __restrict__ bias,
                float* __restrict__ outF,
                unsigned short* __restrict__ outB,
                int N, int O, int I, int headMode, float bscale) {
  int lane = threadIdx.x & 31, wave = threadIdx.x >> 5;
  int tile = blockIdx.x * 8 + wave;
  int tilesO = O >> 6;                 // 64-wide output tiles
  int m0 = (tile / tilesO) << 4;
  int o0 = (tile % tilesO) << 6;
  if (m0 >= N) return;

  v8f acc[4] = {};
  for (int k = 0; k < I; k += 32) {
    Frag16 a;
    load_a_frag(X + (size_t)m0 * I + k, I, lane, a.u);
    Frag16 b0, b1, b2, b3;
    load_bT_frag(W + (size_t)(o0 +  0) * I + k, I, lane, b0.u);
    load_bT_frag(W + (size_t)(o0 + 16) * I + k, I, lane, b1.u);
    load_bT_frag(W + (size_t)(o0 + 32) * I + k, I, lane, b2.u);
    load_bT_frag(W + (size_t)(o0 + 48) * I + k, I, lane, b3.u);
    acc[0] = __builtin_amdgcn_wmma_f32_16x16x32_bf16(false, a.bf, false, b0.bf,
                                                     (short)0, acc[0], false, false);
    acc[1] = __builtin_amdgcn_wmma_f32_16x16x32_bf16(false, a.bf, false, b1.bf,
                                                     (short)0, acc[1], false, false);
    acc[2] = __builtin_amdgcn_wmma_f32_16x16x32_bf16(false, a.bf, false, b2.bf,
                                                     (short)0, acc[2], false, false);
    acc[3] = __builtin_amdgcn_wmma_f32_16x16x32_bf16(false, a.bf, false, b3.bf,
                                                     (short)0, acc[3], false, false);
  }

  int half = lane >> 4, n = lane & 15;
#pragma unroll
  for (int nt = 0; nt < 4; ++nt) {
    int o = o0 + nt * 16 + n;
    float bv = bias ? bias[o] : 0.0f;
#pragma unroll
    for (int i = 0; i < 8; ++i) {
      int m = m0 + half * 8 + i;
      float v = acc[nt][i] + bv;
      if (outF) outF[(size_t)m * O + o] = v;
      if (outB) {
        if (headMode) {
          int bb = m % B_, r = m / B_;
          int h = o >> 6, d = o & 63;
          outB[(((size_t)(bb * H_ + h) * (N / B_) + r) << 6) + d] = f2bf(v * bscale);
        } else {
          outB[(size_t)m * O + o] = f2bf(v * bscale);
        }
      }
    }
  }
}

// ---------------------------------------------------------------------------
// Fused attention for one (b,h) and a 16-query tile.
//   Q [B,H,Lq,64] bf16 (pre-scaled), K/V [B,H,Lk,64] bf16.
// Two-phase softmax using big CDNA5 LDS (full 16 x Lk score strip):
//   phase 1: 8 waves compute S = Q K^T strips via WMMA -> LDS fp32
//   phase 2: 256-thread softmax in LDS, probs re-packed as bf16
//   phase 3: 8 waves compute partial P V via WMMA (A operand from LDS),
//            partials reduced through LDS.
// Outputs: optional bf16 head layout [B,H,Lq,64], optional fp32/bf16 [N,E].
// ---------------------------------------------------------------------------
__global__ void __launch_bounds__(256)
attn_kernel(const unsigned short* __restrict__ Qh,
            const unsigned short* __restrict__ Kh,
            const unsigned short* __restrict__ Vh,
            int Lq, int Lk,
            unsigned short* __restrict__ outHead,
            float* __restrict__ outNE,
            unsigned short* __restrict__ outNEb) {
  extern __shared__ char smem[];
  float*          sS = (float*)smem;                                   // [16][Lk] f32
  unsigned short* sP = (unsigned short*)(smem + (size_t)16 * Lk * 4);  // [16][Lk] bf16
  float*          sO = (float*)(smem + (size_t)16 * Lk * 6);           // [8][16][64] f32
  __shared__ float red[16][16];

  int lane = threadIdx.x & 31, wave = threadIdx.x >> 5;
  int bh = blockIdx.y;
  int q0 = blockIdx.x << 4;

  const unsigned short* Qbase = Qh + ((size_t)bh * Lq + q0) * D_;
  const unsigned short* Kbase = Kh + (size_t)bh * Lk * D_;
  const unsigned short* Vbase = Vh + (size_t)bh * Lk * D_;

  // Q tile fragments (D=64 -> two K-slices of 32), resident across phase 1
  Frag16 aLo, aHi;
  load_a_frag(Qbase + 0,  D_, lane, aLo.u);
  load_a_frag(Qbase + 32, D_, lane, aHi.u);

  // ---- phase 1: scores ----
  int nkt = Lk >> 4;
  for (int kt = wave; kt < nkt; kt += 8) {
    const unsigned short* kb = Kbase + ((size_t)kt << 4) * D_;
    if (kt + 8 < nkt)
      __builtin_prefetch(Kbase + ((size_t)(kt + 8) << 4) * D_, 0, 1);  // global_prefetch
    Frag16 bLo, bHi;
    load_bT_frag(kb + 0,  D_, lane, bLo.u);
    load_bT_frag(kb + 32, D_, lane, bHi.u);
    v8f acc = {};
    acc = __builtin_amdgcn_wmma_f32_16x16x32_bf16(false, aLo.bf, false, bLo.bf,
                                                  (short)0, acc, false, false);
    acc = __builtin_amdgcn_wmma_f32_16x16x32_bf16(false, aHi.bf, false, bHi.bf,
                                                  (short)0, acc, false, false);
    int half = lane >> 4, n = lane & 15;
#pragma unroll
    for (int i = 0; i < 8; ++i) {
      int m = half * 8 + i;
      sS[(size_t)m * Lk + (kt << 4) + n] = acc[i];
    }
  }
  __syncthreads();

  // ---- phase 2: softmax over Lk (row = query) ----
  int row = threadIdx.x >> 4, sub = threadIdx.x & 15;
  float mloc = -3.0e38f;
  for (int k = sub; k < Lk; k += 16) mloc = fmaxf(mloc, sS[(size_t)row * Lk + k]);
  red[row][sub] = mloc;
  __syncthreads();
  float rowm = -3.0e38f;
#pragma unroll
  for (int j = 0; j < 16; ++j) rowm = fmaxf(rowm, red[row][j]);
  float sloc = 0.0f;
  for (int k = sub; k < Lk; k += 16) {
    float e = __expf(sS[(size_t)row * Lk + k] - rowm);
    sS[(size_t)row * Lk + k] = e;
    sloc += e;
  }
  __syncthreads();
  red[row][sub] = sloc;
  __syncthreads();
  float rsum = 0.0f;
#pragma unroll
  for (int j = 0; j < 16; ++j) rsum += red[row][j];
  float rinv = 1.0f / rsum;
  for (int k = sub; k < Lk; k += 16)
    sP[(size_t)row * Lk + k] = f2bf(sS[(size_t)row * Lk + k] * rinv);
  __syncthreads();

  // ---- phase 3: O = P @ V, contraction split across 8 waves ----
  v8f acc0 = {}, acc1 = {}, acc2 = {}, acc3 = {};
  int nkb = Lk >> 5;
  for (int kb = wave; kb < nkb; kb += 8) {
    Frag16 p;
    load_a_frag(sP + (kb << 5), Lk, lane, p.u);  // A operand from LDS (ds loads)
    const unsigned short* vb = Vbase + ((size_t)kb << 5) * D_;
    Frag16 v0, v1, v2, v3;
    load_b_frag(vb + 0,  D_, lane, v0.u);
    load_b_frag(vb + 16, D_, lane, v1.u);
    load_b_frag(vb + 32, D_, lane, v2.u);
    load_b_frag(vb + 48, D_, lane, v3.u);
    acc0 = __builtin_amdgcn_wmma_f32_16x16x32_bf16(false, p.bf, false, v0.bf,
                                                   (short)0, acc0, false, false);
    acc1 = __builtin_amdgcn_wmma_f32_16x16x32_bf16(false, p.bf, false, v1.bf,
                                                   (short)0, acc1, false, false);
    acc2 = __builtin_amdgcn_wmma_f32_16x16x32_bf16(false, p.bf, false, v2.bf,
                                                   (short)0, acc2, false, false);
    acc3 = __builtin_amdgcn_wmma_f32_16x16x32_bf16(false, p.bf, false, v3.bf,
                                                   (short)0, acc3, false, false);
  }
  {
    int half = lane >> 4, n = lane & 15;
    float* myO = sO + (size_t)wave * 16 * 64;
#pragma unroll
    for (int i = 0; i < 8; ++i) {
      int m = half * 8 + i;
      myO[m * 64 + n +  0] = acc0[i];
      myO[m * 64 + n + 16] = acc1[i];
      myO[m * 64 + n + 32] = acc2[i];
      myO[m * 64 + n + 48] = acc3[i];
    }
  }
  __syncthreads();

  // ---- reduce partials + writeout ----
  for (int idx = threadIdx.x; idx < 16 * 64; idx += 256) {
    float v = 0.0f;
#pragma unroll
    for (int w = 0; w < 8; ++w) v += sO[(size_t)w * 1024 + idx];
    int m = idx >> 6, d = idx & 63;
    int q = q0 + m;
    if (outHead) outHead[((size_t)bh * Lq + q) * D_ + d] = f2bf(v);
    if (outNE || outNEb) {
      int bb = bh / H_, h = bh % H_;
      size_t off = (size_t)(q * B_ + bb) * E_ + (h << 6) + d;
      if (outNE)  outNE[off]  = v;
      if (outNEb) outNEb[off] = f2bf(v);
    }
  }
}

// ---------------------------------------------------------------------------
// Gated fusion: attn = g*sumO + (1-g)*regO,  g = sigmoid(G1+G2+gbias)  -> bf16
// ---------------------------------------------------------------------------
__global__ void __launch_bounds__(256)
fuse_gate_kernel(const float* __restrict__ G1, const float* __restrict__ G2,
                 const float* __restrict__ gbias,
                 const float* __restrict__ sumO, const float* __restrict__ regO,
                 unsigned short* __restrict__ attnB) {
  int i = blockIdx.x * 256 + threadIdx.x;
  if (i >= N_ * E_) return;
  int e = i & (E_ - 1);
  float z = G1[i] + G2[i] + gbias[e];
  float g = 1.0f / (1.0f + __expf(-z));
  attnB[i] = f2bf(g * sumO[i] + (1.0f - g) * regO[i]);
}

// ---------------------------------------------------------------------------
extern "C" void kernel_launch(void* const* d_in, const int* in_sizes, int n_in,
                              void* d_out, int out_size, void* d_ws, size_t ws_size,
                              hipStream_t stream) {
  (void)in_sizes; (void)n_in; (void)out_size; (void)ws_size;

  const float* reg_x = (const float*)d_in[0];
  const int*   ids   = (const int*)d_in[1];
  const float* emb   = (const float*)d_in[4];
  const float* wq1   = (const float*)d_in[5];  const float* bq1 = (const float*)d_in[6];
  const float* wq2   = (const float*)d_in[7];  const float* bq2 = (const float*)d_in[8];
  const float* wk    = (const float*)d_in[9];  const float* bk  = (const float*)d_in[10];
  const float* wv    = (const float*)d_in[11]; const float* bv  = (const float*)d_in[12];
  const float* wo    = (const float*)d_in[13]; const float* bo  = (const float*)d_in[14];
  const float* sgw   = (const float*)d_in[15];
  const float* rgw   = (const float*)d_in[16];
  const float* gbias = (const float*)d_in[17];

  // --- workspace carve-up (256B aligned) ---
  char* ws = (char*)d_ws;
  size_t off = 0;
  auto alloc = [&](size_t bytes) -> char* {
    off = (off + 255) & ~(size_t)255;
    char* p = ws + off;
    off += bytes;
    return p;
  };
  const size_t NE = (size_t)N_ * E_;
  const size_t WW = (size_t)E_ * E_;
  const size_t HD = (size_t)B_ * H_ * S_ * D_;

  unsigned short* xbf   = (unsigned short*)alloc(NE * 2);
  unsigned short* wq1b  = (unsigned short*)alloc(WW * 2);
  unsigned short* wq2b  = (unsigned short*)alloc(WW * 2);
  unsigned short* wkb   = (unsigned short*)alloc(WW * 2);
  unsigned short* wvb   = (unsigned short*)alloc(WW * 2);
  unsigned short* wob   = (unsigned short*)alloc(WW * 2);
  unsigned short* sgwb  = (unsigned short*)alloc(WW * 2);
  unsigned short* rgwb  = (unsigned short*)alloc(WW * 2);
  unsigned short* Khd   = (unsigned short*)alloc(NE * 2);   // [B,H,R,64]
  unsigned short* Vhd   = (unsigned short*)alloc(NE * 2);
  unsigned short* Q1h   = (unsigned short*)alloc(NE * 2);
  unsigned short* Q2h   = (unsigned short*)alloc(NE * 2);
  unsigned short* sumQ  = (unsigned short*)alloc(HD * 2);   // [B,H,S,64]
  unsigned short* sumX2 = (unsigned short*)alloc(HD * 2);
  float*          sumOF = (float*)alloc(NE * 4);
  unsigned short* sumOB = (unsigned short*)alloc(NE * 2);
  float*          regOF = (float*)alloc(NE * 4);
  unsigned short* regOB = (unsigned short*)alloc(NE * 2);
  float*          G1    = (float*)alloc(NE * 4);
  float*          G2    = (float*)alloc(NE * 4);
  unsigned short* attnB = (unsigned short*)alloc(NE * 2);

  const int cvtNE = (int)((NE + 255) / 256);
  const int cvtWW = (int)((WW + 255) / 256);
  const int gemmBlocks = (N_ / 16) * (E_ / 64) / 8;   // 256 blocks, 16x64 tile/wave

  // 1) bf16 conversions
  f32_to_bf16_kernel<<<cvtNE, 256, 0, stream>>>(reg_x, xbf, (int)NE);
  f32_to_bf16_kernel<<<cvtWW, 256, 0, stream>>>(wq1, wq1b, (int)WW);
  f32_to_bf16_kernel<<<cvtWW, 256, 0, stream>>>(wq2, wq2b, (int)WW);
  f32_to_bf16_kernel<<<cvtWW, 256, 0, stream>>>(wk,  wkb,  (int)WW);
  f32_to_bf16_kernel<<<cvtWW, 256, 0, stream>>>(wv,  wvb,  (int)WW);
  f32_to_bf16_kernel<<<cvtWW, 256, 0, stream>>>(wo,  wob,  (int)WW);
  f32_to_bf16_kernel<<<cvtWW, 256, 0, stream>>>(sgw, sgwb, (int)WW);
  f32_to_bf16_kernel<<<cvtWW, 256, 0, stream>>>(rgw, rgwb, (int)WW);

  // 2) summary queries (embedding gather, pre-scaled)
  gather_sumq_kernel<<<(int)((HD + 255) / 256), 256, 0, stream>>>(emb, ids, sumQ);

  // 3) projections -> head-layout bf16 (Q pre-scaled by 1/sqrt(D))
  gemm_xwT_kernel<<<gemmBlocks, 256, 0, stream>>>(xbf, wkb,  bk,  nullptr, Khd, N_, E_, E_, 1, 1.0f);
  gemm_xwT_kernel<<<gemmBlocks, 256, 0, stream>>>(xbf, wvb,  bv,  nullptr, Vhd, N_, E_, E_, 1, 1.0f);
  gemm_xwT_kernel<<<gemmBlocks, 256, 0, stream>>>(xbf, wq1b, bq1, nullptr, Q1h, N_, E_, E_, 1, SCALE_);
  gemm_xwT_kernel<<<gemmBlocks, 256, 0, stream>>>(xbf, wq2b, bq2, nullptr, Q2h, N_, E_, E_, 1, SCALE_);

  auto shmem = [](int Lk) -> size_t { return (size_t)16 * Lk * 6 + 8 * 16 * 64 * 4; };

  // 4) stage 1: summary queries attend to regular tokens -> sum_x2 (bf16 head)
  attn_kernel<<<dim3(S_ / 16, B_ * H_), 256, shmem(R_), stream>>>(
      sumQ, Khd, Vhd, S_, R_, sumX2, nullptr, nullptr);

  // 5) stage 2: regular queries attend to summary tokens (K=V=sum_x2)
  attn_kernel<<<dim3(R_ / 16, B_ * H_), 256, shmem(S_), stream>>>(
      Q1h, sumX2, sumX2, R_, S_, nullptr, sumOF, sumOB);

  // 6) stage 3: regular self-attention
  attn_kernel<<<dim3(R_ / 16, B_ * H_), 256, shmem(R_), stream>>>(
      Q2h, Khd, Vhd, R_, R_, nullptr, regOF, regOB);

  // 7) gate GEMMs + fusion
  gemm_xwT_kernel<<<gemmBlocks, 256, 0, stream>>>(sumOB, sgwb, nullptr, G1, nullptr, N_, E_, E_, 0, 1.0f);
  gemm_xwT_kernel<<<gemmBlocks, 256, 0, stream>>>(regOB, rgwb, nullptr, G2, nullptr, N_, E_, E_, 0, 1.0f);
  fuse_gate_kernel<<<cvtNE, 256, 0, stream>>>(G1, G2, gbias, sumOF, regOF, attnB);

  // 8) output projection -> d_out [R,B,E] fp32
  gemm_xwT_kernel<<<gemmBlocks, 256, 0, stream>>>(attnB, wob, bo, (float*)d_out, nullptr, N_, E_, E_, 0, 1.0f);
}
